// BaseMultiLora_63883343560842
// MI455X (gfx1250) — matrix operand order repaired and checked by reading, not compiled, not run on hardware
//
#include <hip/hip_runtime.h>

// Problem constants (from reference)
#define B_  32
#define S_  2048
#define H_  4096
#define R_  64
#define NA_ 16

typedef float        f32x8   __attribute__((ext_vector_type(8)));
typedef float        v8f     __attribute__((ext_vector_type(8)));
typedef __bf16       bf16x16 __attribute__((ext_vector_type(16)));

union BF16x16 {
    unsigned u[8];
    bf16x16  v;
};

static __device__ __forceinline__ unsigned fb(float f)   { return __builtin_bit_cast(unsigned, f); }
static __device__ __forceinline__ float    bf(unsigned u){ return __builtin_bit_cast(float, u); }

// Pack the high 16 bits of two fp32 into one dword: {u1[31:16], u0[31:16]}
// via a single v_perm_b32 (result byte i <- sel byte i; 0-3 from S1(=b), 4-7 from S0(=a)).
static __device__ __forceinline__ unsigned pack_hi16(unsigned u1, unsigned u0) {
    return __builtin_amdgcn_perm(u1, u0, 0x07060302u);
}

// Split 2 fp32 -> packed bf16 hi (truncation) + packed bf16 lo (exact residual, truncated).
// hi+lo reproduces fp32 to ~2^-16 relative.
static __device__ __forceinline__ void split2(float e0, float e1, unsigned& hi, unsigned& lo) {
    unsigned u0 = fb(e0), u1 = fb(e1);
    hi = pack_hi16(u1, u0);
    float l0 = e0 - bf(u0 & 0xFFFF0000u);   // lowers to v_pk_add_f32 w/ neg
    float l1 = e1 - bf(u1 & 0xFFFF0000u);
    lo = pack_hi16(fb(l1), fb(l0));
}

// Split 8 fp32 (K-ascending) into 4 hi dwords + 4 lo dwords.
static __device__ __forceinline__ void split_f32x8(f32x8 vsrc, unsigned* hi, unsigned* lo) {
#pragma unroll
    for (int j = 0; j < 4; ++j)
        split2(vsrc[2 * j], vsrc[2 * j + 1], hi[j], lo[j]);
}

// Prepass: split the (tiny, heavily reused, L2-resident) weight tensor into bf16 hi/lo planes.
__global__ void __launch_bounds__(256)
split_w_kernel(const float* __restrict__ w,
               unsigned* __restrict__ whi,
               unsigned* __restrict__ wlo) {
    int i = blockIdx.x * 256 + threadIdx.x;          // fp32 pair index
    float e0 = w[2 * i], e1 = w[2 * i + 1];
    unsigned hi, lo;
    split2(e0, e1, hi, lo);
    whi[i] = hi;
    wlo[i] = lo;
}

// Main GEMM. Each wave owns a 16-row M-tile and computes ALL of R=64 (four 16x16
// N-tiles, 4 accumulators), so the fp32->bf16 split of A feeds 12 WMMAs per chunk.
// bf16x3: hi*hi + hi*lo + lo*hi, fp32 accumulate in v_wmma_f32_16x16x32_bf16.
template <bool PREW>
__global__ void __launch_bounds__(128)
lora_wmma_kernel(const float* __restrict__ x,
                 const int* __restrict__ ids,
                 const unsigned short* __restrict__ whi16,
                 const unsigned short* __restrict__ wlo16,
                 const float* __restrict__ wfp,
                 float* __restrict__ out) {
    const int lane    = threadIdx.x & 31;
    const int wave    = threadIdx.x >> 5;
    const int rowbase = (blockIdx.x * 4 + wave) * 16;  // flattened over B*S
    const int b       = rowbase >> 11;                 // / S_ (2048)
    const int aid     = ids[b];
    const int m       = lane & 15;                     // A row / B col / C col within tile
    const int h       = lane >> 4;                     // half-wave select

    // A: row (rowbase+m); elems 0..15 <- 16 contiguous K at h*16
    const float* ap = x + (size_t)(rowbase + m) * H_ + h * 16;

    // B: column (ntile*16+m); elems 0..15 <- 16 contiguous K at h*16.
    // ntile offsets (n*16*H elements) fold into the 24-bit instruction offset.
    const size_t wrow = ((size_t)aid * R_ + m) * H_ + h * 16;
    const unsigned short* bhp = nullptr;
    const unsigned short* blp = nullptr;
    const float*          bfp = nullptr;
    if (PREW) {
        bhp = whi16 + wrow;
        blp = wlo16 + wrow;
    } else {
        bfp = wfp + wrow;
    }

    v8f acc0 = {}, acc1 = {}, acc2 = {}, acc3 = {};
    v8f* acc[4] = {&acc0, &acc1, &acc2, &acc3};

#pragma unroll 2
    for (int k = 0; k < H_; k += 32) {
        f32x8 a0 = *(const f32x8*)(ap);                // K: h*16 .. h*16+7
        f32x8 a1 = *(const f32x8*)(ap + 8);            // K: h*16+8 .. h*16+15
        ap += 32;

        BF16x16 ah, al;
        split_f32x8(a0, ah.u,     al.u);
        split_f32x8(a1, ah.u + 4, al.u + 4);

#pragma unroll
        for (int n = 0; n < 4; ++n) {
            BF16x16 bh, bl;
            if (PREW) {
                bh.v = *(const bf16x16*)(bhp + (size_t)n * 16 * H_);
                bl.v = *(const bf16x16*)(blp + (size_t)n * 16 * H_);
            } else {
                f32x8 b0 = *(const f32x8*)(bfp + (size_t)n * 16 * H_);
                f32x8 b1 = *(const f32x8*)(bfp + (size_t)n * 16 * H_ + 8);
                split_f32x8(b0, bh.u,     bl.u);
                split_f32x8(b1, bh.u + 4, bl.u + 4);
            }
            v8f a = *acc[n];
            a = __builtin_amdgcn_wmma_f32_16x16x32_bf16(false, ah.v, false, bh.v,
                                                        (short)0, a, false, false);
            a = __builtin_amdgcn_wmma_f32_16x16x32_bf16(false, ah.v, false, bl.v,
                                                        (short)0, a, false, false);
            a = __builtin_amdgcn_wmma_f32_16x16x32_bf16(false, al.v, false, bh.v,
                                                        (short)0, a, false, false);
            *acc[n] = a;
        }
        if (PREW) { bhp += 32; blp += 32; }
        else      { bfp += 32; }
    }

    // C/D layout: VGPR i -> M = i + 8*h, N = m
#pragma unroll
    for (int n = 0; n < 4; ++n) {
        float* op = out + (size_t)(rowbase + h * 8) * R_ + n * 16 + m;
        v8f a = *acc[n];
#pragma unroll
        for (int i = 0; i < 8; ++i) {
            *op = a[i];
            op += R_;
        }
    }
}

extern "C" void kernel_launch(void* const* d_in, const int* in_sizes, int n_in,
                              void* d_out, int out_size, void* d_ws, size_t ws_size,
                              hipStream_t stream) {
    const float* x   = (const float*)d_in[0];
    const int*   ids = (const int*)d_in[1];
    const float* w   = (const float*)d_in[2];
    float*       out = (float*)d_out;

    const size_t wcnt = (size_t)NA_ * R_ * H_;               // 4,194,304 elements
    const size_t need = wcnt * 2 * sizeof(unsigned short);   // hi+lo bf16 planes (16 MB)

    const int nblocks = (B_ * S_) / 64;                      // 1024 blocks, 4 waves each

    if (ws_size >= need) {
        unsigned* whi = (unsigned*)d_ws;                     // wcnt bf16 as wcnt/2 dwords
        unsigned* wlo = whi + wcnt / 2;
        const int pre_blocks = (int)(wcnt / 2 / 256);
        split_w_kernel<<<pre_blocks, 256, 0, stream>>>(w, whi, wlo);
        lora_wmma_kernel<true><<<nblocks, 128, 0, stream>>>(
            x, ids, (const unsigned short*)whi, (const unsigned short*)wlo, nullptr, out);
    } else {
        lora_wmma_kernel<false><<<nblocks, 128, 0, stream>>>(
            x, ids, nullptr, nullptr, w, out);
    }
}
